// MFN3_1271310319775
// MI455X (gfx1250) — matrix operand order repaired and checked by reading, not compiled, not run on hardware
//
#include <hip/hip_runtime.h>
#include <math.h>

typedef __attribute__((ext_vector_type(16))) _Float16 v16h;
typedef __attribute__((ext_vector_type(8)))  float    v8f;

union V16H { v16h v; unsigned int u[8]; _Float16 h[16]; };

#define GLOBAL_AS __attribute__((address_space(1)))

// Force global (addrspace 1) access so the compiler emits global_load/store
// (LOADcnt/STOREcnt only) instead of flat_* (which also couples to DScnt).
template <typename T>
__device__ __forceinline__ T gload(const T* p) {
  return *(const GLOBAL_AS T*)(unsigned long long)p;
}
template <typename T>
__device__ __forceinline__ void gstore(T* p, T v) {
  *(GLOBAL_AS T*)(unsigned long long)p = v;
}

__device__ inline float sigmoidf_(float x) { return 1.f / (1.f + __expf(-x)); }

// ---------------------------------------------------------------------------
// WMMA helpers (CDNA5 gfx1250, wave32)
// ---------------------------------------------------------------------------
__device__ inline v8f wmma16(v16h a, v16h b, v8f c) {
  return __builtin_amdgcn_wmma_f32_16x16x32_f16(false, a, false, b, (short)0, c,
                                                false, false);
}

// A-operand gather from an f16 LDS tile, per ISA 16-bit A 16x32 layout:
// lane L: m = L&15, half = L>>4; VGPR v holds K pairs at
// k0 = (v<4 ? 2v : 16+2(v-4)) + 8*half  (low half = k0, high half = k0+1)
// The two 4-dword groups are contiguous -> compiler merges to 2x ds_load_b128.
__device__ inline v16h load_A_f16(const _Float16* base, int stride, int colOff,
                                  int lane) {
  int m = lane & 15, half = lane >> 4;
  const _Float16* row = base + m * stride + colOff + 8 * half;
  V16H r;
#pragma unroll
  for (int v = 0; v < 8; ++v) {
    int k0 = (v < 4) ? (2 * v) : (16 + 2 * (v - 4));
    r.u[v] = *(const unsigned int*)(row + k0);
  }
  return r.v;
}

// B-operand load from pre-packed weights: tile (kt,nt) = 512 halves,
// lane-major (lane owns 16 contiguous halves => 2x global_load_b128).
__device__ inline v16h load_B_packed(const _Float16* __restrict__ packed,
                                     int kt, int nt, int KT, int lane) {
  const _Float16* p = packed + ((size_t)(nt * KT + kt) * 512) + lane * 16;
  return gload((const v16h*)p);
}

__device__ inline v8f gemm_lds(const _Float16* A, int astride, int aoff,
                               const _Float16* __restrict__ Bp, int KT, int nt,
                               int lane) {
  v8f acc = {};
  for (int kt = 0; kt < KT; ++kt) {
    v16h a = load_A_f16(A, astride, aoff + kt * 32, lane);
    v16h b = load_B_packed(Bp, kt, nt, KT, lane);
    acc = wmma16(a, b, acc);
  }
  return acc;
}

// C/D layout: lane L, vgpr r -> m = r + 8*(L>>4), n = L&15.
// ACT: 0=none, 1=relu, 2=sigmoid, 3=tanh
template <int ACT>
__device__ inline void store_tile(_Float16* dst, int dstride, int coff, v8f acc,
                                  float bv, int lane) {
  int n = lane & 15, half = lane >> 4;
#pragma unroll
  for (int r = 0; r < 8; ++r) {
    int m = r + 8 * half;
    float x = acc[r] + bv;
    if (ACT == 1) x = fmaxf(x, 0.f);
    else if (ACT == 2) x = sigmoidf_(x);
    else if (ACT == 3) x = tanhf(x);
    dst[m * dstride + coff + n] = (_Float16)x;
  }
}

// ---------------------------------------------------------------------------
// Pack f32 row-major [K,N] weight into WMMA-B f16 tile layout.
// Tile (kt,nt): lane L owns halves e=0..15: j=e>>1, p=e&1,
//   k = kt*32 + (e<8 ? e + 8*(L>>4) : 16 + (e-8) + 8*(L>>4)),  n = nt*16 + (L&15). Zero-pad k>=K.
// ---------------------------------------------------------------------------
__global__ void pack_kernel(const float* __restrict__ src,
                            _Float16* __restrict__ dst, int K, int N, int KT,
                            int NT) {
  int tile = blockIdx.x;
  int kt = tile % KT, nt = tile / KT;
  int lane = threadIdx.x;
  int half = lane >> 4;
  int n = nt * 16 + (lane & 15);
  _Float16* out = dst + (size_t)tile * 512 + lane * 16;
#pragma unroll
  for (int e = 0; e < 16; ++e) {
    int j = e >> 1, p = e & 1;
    int k = kt * 32 + (e < 8 ? (2 * j + p + 8 * half) : (16 + 2 * j + p - 8 + 8 * half));
    float v = (k < K) ? gload(&src[(size_t)k * N + n]) : 0.f;
    gstore(&out[e], (_Float16)v);
  }
}

// ---------------------------------------------------------------------------
// Input projection: G[t][b][0:768] = [x_l@Wl | x_a@Wa | x_v@Wv]  (f16 out)
// x rows are bt = b*200 + t; G is stored [T,B,768] for contiguous reads later.
// Block = 256 threads (8 waves), one 16-row M tile per block; each wave owns
// 2 N-tiles per modality; A staged in LDS as f16 in 128-col K chunks.
// ---------------------------------------------------------------------------
__global__ __launch_bounds__(256) void proj_kernel(
    const float* __restrict__ x, const _Float16* __restrict__ pWl,
    const _Float16* __restrict__ pWa, const _Float16* __restrict__ pWv,
    _Float16* __restrict__ G) {
  __shared__ alignas(16) _Float16 xs[16][128];
  const int tid = threadIdx.x, lane = tid & 31, wid = tid >> 5;
  const int mtile = blockIdx.x, row0 = mtile * 16;
  const int DMOD[3] = {711, 300, 74};
  const int KTm[3] = {23, 10, 3};
  const int CB[3] = {0, 711, 1011};
  const _Float16* PW[3] = {pWl, pWa, pWv};

  for (int mod = 0; mod < 3; ++mod) {
    v8f acc0 = {}, acc1 = {};
    const int KT = KTm[mod], D = DMOD[mod], cb = CB[mod];
    const int nchunks = (KT + 3) / 4;
    for (int ch = 0; ch < nchunks; ++ch) {
      __syncthreads();
#pragma unroll
      for (int i = 0; i < 8; ++i) {
        int flat = tid + i * 256;  // 16*128 = 2048
        int r = flat >> 7, cc = flat & 127;
        int cmod = ch * 128 + cc;
        float v =
            (cmod < D) ? gload(&x[(size_t)(row0 + r) * 1085 + cb + cmod]) : 0.f;
        xs[r][cc] = (_Float16)v;
      }
      __syncthreads();
      for (int kl = 0; kl < 4; ++kl) {
        int kt = ch * 4 + kl;
        if (kt >= KT) break;
        v16h a = load_A_f16(&xs[0][0], 128, kl * 32, lane);
        v16h b0 = load_B_packed(PW[mod], kt, 2 * wid, KT, lane);
        acc0 = wmma16(a, b0, acc0);
        v16h b1 = load_B_packed(PW[mod], kt, 2 * wid + 1, KT, lane);
        acc1 = wmma16(a, b1, acc1);
      }
    }
    int n = lane & 15, half = lane >> 4;
#pragma unroll
    for (int r = 0; r < 8; ++r) {
      int m = r + 8 * half;
      int gr = row0 + m;
      int b = gr / 200, t = gr % 200;
      size_t base = ((size_t)(t * 256 + b)) * 768 + mod * 256;
      gstore(&G[base + (2 * wid) * 16 + n], (_Float16)acc0[r]);
      gstore(&G[base + (2 * wid + 1) * 16 + n], (_Float16)acc1[r]);
    }
  }
}

// ---------------------------------------------------------------------------
// Recurrent kernel: 16 blocks, each owns 16 batch rows for all 200 steps.
// All state in LDS; all matmuls via WMMA (M=16 tiles); weights stream from L2.
// ---------------------------------------------------------------------------
__global__ __launch_bounds__(256) void recurrent_kernel(
    const _Float16* __restrict__ G, const _Float16* __restrict__ pWhhL,
    const _Float16* __restrict__ pWhhA, const _Float16* __restrict__ pWhhV,
    const _Float16* __restrict__ pA1W1, const float* __restrict__ a1b1,
    const _Float16* __restrict__ pA1W2, const float* __restrict__ a1b2,
    const _Float16* __restrict__ pA2W1, const float* __restrict__ a2b1,
    const _Float16* __restrict__ pA2W2, const float* __restrict__ a2b2,
    const _Float16* __restrict__ pG1W1, const float* __restrict__ g1b1,
    const _Float16* __restrict__ pG1W2, const float* __restrict__ g1b2,
    const _Float16* __restrict__ pG2W1, const float* __restrict__ g2b1,
    const _Float16* __restrict__ pG2W2, const float* __restrict__ g2b2,
    const _Float16* __restrict__ pOW1, const float* __restrict__ ow2,
    float* __restrict__ out) {
  __shared__ float h_s[3][16][64];
  __shared__ float c_s[3][16][64];
  __shared__ float mem_s[16][64];
  __shared__ alignas(16) _Float16 bothF16[16][448];  // [cstar(384)|mem(64)] f16
  __shared__ alignas(16) _Float16 hF16[16][192];     // staged h / MLP hiddens
  __shared__ alignas(16) _Float16 scrF16[16][384];   // scores, then chat|g1|g2
  __shared__ float red[16][16];
  __shared__ float rowv[16];

  const int tid = threadIdx.x, lane = tid & 31, wid = tid >> 5;
  const int half = lane >> 4, ln = lane & 15;
  const int b0 = blockIdx.x * 16;
  const _Float16* pWhh[3] = {pWhhL, pWhhA, pWhhV};

  for (int i = tid; i < 3 * 16 * 64; i += 256) {
    ((float*)h_s)[i] = 0.f;
    ((float*)c_s)[i] = 0.f;
  }
  for (int i = tid; i < 16 * 64; i += 256) ((float*)mem_s)[i] = 0.f;
  __syncthreads();

  for (int t = 0; t < 200; ++t) {
    // stage h (all 3 modalities) and mem as f16 A-operands
    for (int i = tid; i < 16 * 192; i += 256) {
      int r = i / 192, cc = i % 192;
      hF16[r][cc] = (_Float16)h_s[cc / 64][r][cc & 63];
    }
    for (int i = tid; i < 16 * 64; i += 256) {
      int r = i >> 6, k = i & 63;
      bothF16[r][384 + k] = (_Float16)mem_s[r][k];
    }
    __syncthreads();

    // LSTM gates: 12 jobs = (modality, 16-col tile); acc init from G (x@Wih)
    for (int job = wid; job < 12; job += 8) {
      int mod = job >> 2, ct = job & 3;
      v8f acc[4];
#pragma unroll
      for (int gt = 0; gt < 4; ++gt) {
        v8f z = {};
        acc[gt] = z;
#pragma unroll
        for (int r = 0; r < 8; ++r) {
          int m = r + 8 * half;
          size_t gi = ((size_t)(t * 256 + b0 + m)) * 768 + mod * 256 + gt * 64 +
                      ct * 16 + ln;
          acc[gt][r] = (float)gload(&G[gi]);
        }
      }
#pragma unroll
      for (int kt = 0; kt < 2; ++kt) {
        v16h a = load_A_f16(&hF16[0][0], 192, mod * 64 + kt * 32, lane);
#pragma unroll
        for (int gt = 0; gt < 4; ++gt) {
          v16h bm = load_B_packed(pWhh[mod], kt, gt * 4 + ct, 2, lane);
          acc[gt] = wmma16(a, bm, acc[gt]);
        }
      }
#pragma unroll
      for (int r = 0; r < 8; ++r) {
        int m = r + 8 * half;
        int col = ct * 16 + ln;
        float iv = sigmoidf_(acc[0][r]);
        float fv = sigmoidf_(acc[1][r]);
        float gv = tanhf(acc[2][r]);
        float ov = sigmoidf_(acc[3][r]);
        float cold = c_s[mod][m][col];
        bothF16[m][mod * 64 + col] = (_Float16)cold;       // c_star old
        float cn = fv * cold + iv * gv;
        c_s[mod][m][col] = cn;
        bothF16[m][192 + mod * 64 + col] = (_Float16)cn;   // c_star new
        h_s[mod][m][col] = ov * tanhf(cn);
      }
    }
    __syncthreads();

    // att1 layer1: relu(c_star @ att1_w1 + b1) -> hF16[:, :64]
    if (wid < 4) {
      v8f acc = gemm_lds(&bothF16[0][0], 448, 0, pA1W1, 12, wid, lane);
      store_tile<1>(&hF16[0][0], 192, wid * 16, acc, gload(&a1b1[wid * 16 + ln]),
                    lane);
    }
    __syncthreads();

    // att1 layer2: scores = hidden @ att1_w2 + b2 -> scrF16 (384 wide)
    for (int j = 0; j < 3; ++j) {
      int nt = wid * 3 + j;
      v8f acc = gemm_lds(&hF16[0][0], 192, 0, pA1W2, 2, nt, lane);
      store_tile<0>(&scrF16[0][0], 384, nt * 16, acc, gload(&a1b2[nt * 16 + ln]),
                    lane);
    }
    __syncthreads();

    // softmax over 384 + attended = attention * c_star (in-place on bothF16)
    {
      int r = tid >> 4, s = tid & 15;
      float lmax = -1e30f;
      for (int j = 0; j < 24; ++j)
        lmax = fmaxf(lmax, (float)scrF16[r][s + j * 16]);
      red[r][s] = lmax;
      __syncthreads();
      if (s == 0) {
        float m = red[r][0];
        for (int j = 1; j < 16; ++j) m = fmaxf(m, red[r][j]);
        rowv[r] = m;
      }
      __syncthreads();
      float rmax = rowv[r];
      float lsum = 0.f;
      for (int j = 0; j < 24; ++j)
        lsum += __expf((float)scrF16[r][s + j * 16] - rmax);
      red[r][s] = lsum;
      __syncthreads();
      if (s == 0) {
        float m = 0.f;
        for (int j = 0; j < 16; ++j) m += red[r][j];
        rowv[r] = m;
      }
      __syncthreads();
      float inv = 1.f / rowv[r];
      for (int j = 0; j < 24; ++j) {
        int cidx = s + j * 16;
        float att = __expf((float)scrF16[r][cidx] - rmax) * inv;
        bothF16[r][cidx] = (_Float16)(att * (float)bothF16[r][cidx]);
      }
    }
    __syncthreads();

    // layer1 of att2 / gamma1 / gamma2 (12 jobs) -> hF16 blocks [0|64|128]
    for (int job = wid; job < 12; job += 8) {
      int which = job >> 2, nt = job & 3;
      v8f acc;
      const float* bias;
      if (which == 0) {
        acc = gemm_lds(&bothF16[0][0], 448, 0, pA2W1, 12, nt, lane);
        bias = a2b1;
      } else if (which == 1) {
        acc = gemm_lds(&bothF16[0][0], 448, 0, pG1W1, 14, nt, lane);
        bias = g1b1;
      } else {
        acc = gemm_lds(&bothF16[0][0], 448, 0, pG2W1, 14, nt, lane);
        bias = g2b1;
      }
      store_tile<1>(&hF16[0][0], 192, which * 64 + nt * 16, acc,
                    gload(&bias[nt * 16 + ln]), lane);
    }
    __syncthreads();

    // layer2: chat(tanh) | gamma1(sig) | gamma2(sig) -> scrF16 [0|64|128]
    for (int job = wid; job < 12; job += 8) {
      int which = job >> 2, nt = job & 3;
      if (which == 0) {
        v8f acc = gemm_lds(&hF16[0][0], 192, 0, pA2W2, 2, nt, lane);
        store_tile<3>(&scrF16[0][0], 384, nt * 16, acc,
                      gload(&a2b2[nt * 16 + ln]), lane);
      } else if (which == 1) {
        v8f acc = gemm_lds(&hF16[0][0], 192, 64, pG1W2, 2, nt, lane);
        store_tile<2>(&scrF16[0][0], 384, 64 + nt * 16, acc,
                      gload(&g1b2[nt * 16 + ln]), lane);
      } else {
        v8f acc = gemm_lds(&hF16[0][0], 192, 128, pG2W2, 2, nt, lane);
        store_tile<2>(&scrF16[0][0], 384, 128 + nt * 16, acc,
                      gload(&g2b2[nt * 16 + ln]), lane);
      }
    }
    __syncthreads();

    // mem = gamma1*mem + gamma2*chat
    for (int i = tid; i < 16 * 64; i += 256) {
      int r = i >> 6, k = i & 63;
      float g1v = (float)scrF16[r][64 + k];
      float g2v = (float)scrF16[r][128 + k];
      float ch = (float)scrF16[r][k];
      mem_s[r][k] = g1v * mem_s[r][k] + g2v * ch;
    }
    __syncthreads();
  }

  // readout: relu([h_l|h_a|h_v|mem] @ out_w1) @ out_w2
  for (int i = tid; i < 16 * 256; i += 256) {
    int r = i >> 8, cc = i & 255;
    int blkc = cc >> 6, k = cc & 63;
    float v = (blkc < 3) ? h_s[blkc][r][k] : mem_s[r][k];
    bothF16[r][cc] = (_Float16)v;
  }
  __syncthreads();
  if (wid < 4) {
    v8f acc = gemm_lds(&bothF16[0][0], 448, 0, pOW1, 8, wid, lane);
    store_tile<1>(&hF16[0][0], 192, wid * 16, acc, 0.f, lane);
  }
  __syncthreads();
  if (tid < 16) {
    float s = 0.f;
    for (int k = 0; k < 64; ++k) s += (float)hF16[tid][k] * gload(&ow2[k]);
    gstore(&out[b0 + tid], s);
  }
}

// ---------------------------------------------------------------------------
extern "C" void kernel_launch(void* const* d_in, const int* in_sizes, int n_in,
                              void* d_out, int out_size, void* d_ws,
                              size_t ws_size, hipStream_t stream) {
  (void)in_sizes; (void)n_in; (void)out_size; (void)ws_size;
  const float* x = (const float*)d_in[0];
  _Float16* ws = (_Float16*)d_ws;

  constexpr size_t G_HALVES = (size_t)200 * 256 * 768;
  constexpr size_t SZ_WIHL = (size_t)23 * 16 * 512;
  constexpr size_t SZ_WIHA = (size_t)10 * 16 * 512;
  constexpr size_t SZ_WIHV = (size_t)3 * 16 * 512;
  constexpr size_t SZ_WHH = (size_t)2 * 16 * 512;
  constexpr size_t SZ_A1W1 = (size_t)12 * 4 * 512;
  constexpr size_t SZ_A1W2 = (size_t)2 * 24 * 512;
  constexpr size_t SZ_A2W1 = (size_t)12 * 4 * 512;
  constexpr size_t SZ_SM = (size_t)2 * 4 * 512;
  constexpr size_t SZ_GW1 = (size_t)14 * 4 * 512;
  constexpr size_t SZ_OW1 = (size_t)8 * 4 * 512;

  _Float16* G = ws;
  size_t o = G_HALVES;
  _Float16* pWihL = ws + o; o += SZ_WIHL;
  _Float16* pWihA = ws + o; o += SZ_WIHA;
  _Float16* pWihV = ws + o; o += SZ_WIHV;
  _Float16* pWhhL = ws + o; o += SZ_WHH;
  _Float16* pWhhA = ws + o; o += SZ_WHH;
  _Float16* pWhhV = ws + o; o += SZ_WHH;
  _Float16* pA1W1 = ws + o; o += SZ_A1W1;
  _Float16* pA1W2 = ws + o; o += SZ_A1W2;
  _Float16* pA2W1 = ws + o; o += SZ_A2W1;
  _Float16* pA2W2 = ws + o; o += SZ_SM;
  _Float16* pG1W1 = ws + o; o += SZ_GW1;
  _Float16* pG1W2 = ws + o; o += SZ_SM;
  _Float16* pG2W1 = ws + o; o += SZ_GW1;
  _Float16* pG2W2 = ws + o; o += SZ_SM;
  _Float16* pOW1 = ws + o; o += SZ_OW1;

  pack_kernel<<<23 * 16, 32, 0, stream>>>((const float*)d_in[1], pWihL, 711, 256, 23, 16);
  pack_kernel<<<10 * 16, 32, 0, stream>>>((const float*)d_in[3], pWihA, 300, 256, 10, 16);
  pack_kernel<<<3 * 16, 32, 0, stream>>>((const float*)d_in[5], pWihV, 74, 256, 3, 16);
  pack_kernel<<<2 * 16, 32, 0, stream>>>((const float*)d_in[2], pWhhL, 64, 256, 2, 16);
  pack_kernel<<<2 * 16, 32, 0, stream>>>((const float*)d_in[4], pWhhA, 64, 256, 2, 16);
  pack_kernel<<<2 * 16, 32, 0, stream>>>((const float*)d_in[6], pWhhV, 64, 256, 2, 16);
  pack_kernel<<<12 * 4, 32, 0, stream>>>((const float*)d_in[7], pA1W1, 384, 64, 12, 4);
  pack_kernel<<<2 * 24, 32, 0, stream>>>((const float*)d_in[9], pA1W2, 64, 384, 2, 24);
  pack_kernel<<<12 * 4, 32, 0, stream>>>((const float*)d_in[11], pA2W1, 384, 64, 12, 4);
  pack_kernel<<<2 * 4, 32, 0, stream>>>((const float*)d_in[13], pA2W2, 64, 64, 2, 4);
  pack_kernel<<<14 * 4, 32, 0, stream>>>((const float*)d_in[15], pG1W1, 448, 64, 14, 4);
  pack_kernel<<<2 * 4, 32, 0, stream>>>((const float*)d_in[17], pG1W2, 64, 64, 2, 4);
  pack_kernel<<<14 * 4, 32, 0, stream>>>((const float*)d_in[19], pG2W1, 448, 64, 14, 4);
  pack_kernel<<<2 * 4, 32, 0, stream>>>((const float*)d_in[21], pG2W2, 64, 64, 2, 4);
  pack_kernel<<<8 * 4, 32, 0, stream>>>((const float*)d_in[23], pOW1, 256, 64, 8, 4);

  proj_kernel<<<3200, 256, 0, stream>>>(x, pWihL, pWihA, pWihV, G);

  recurrent_kernel<<<16, 256, 0, stream>>>(
      G, pWhhL, pWhhA, pWhhV, pA1W1, (const float*)d_in[8], pA1W2,
      (const float*)d_in[10], pA2W1, (const float*)d_in[12], pA2W2,
      (const float*)d_in[14], pG1W1, (const float*)d_in[16], pG1W2,
      (const float*)d_in[18], pG2W1, (const float*)d_in[20], pG2W2,
      (const float*)d_in[22], pOW1, (const float*)d_in[24], (float*)d_out);
}
